// TransformerBlock_11605001634069
// MI455X (gfx1250) — matrix operand order repaired and checked by reading, not compiled, not run on hardware
//
#include <hip/hip_runtime.h>

// ---------------------------------------------------------------------------
// Types for CDNA5 WMMA (wave32, 16x16x32 bf16 -> f32)
// ---------------------------------------------------------------------------
typedef __attribute__((ext_vector_type(16))) __bf16 v16bf;
typedef __attribute__((ext_vector_type(8)))  float  v8f;
typedef __attribute__((ext_vector_type(4)))  int    v4i;
typedef __bf16 bf16_t;

constexpr int EPI_BIAS    = 1;
constexpr int EPI_RELU    = 2;
constexpr int EPI_RESID   = 4;
constexpr int EPI_OUTBF16 = 8;
constexpr int EPI_TRANSC  = 16;   // store C transposed: C[col][row]

// ---------------------------------------------------------------------------
// CDNA5 async global->LDS copy path (guarded: falls back to sync loads)
// ---------------------------------------------------------------------------
#if defined(__HIP_DEVICE_COMPILE__) && __has_builtin(__builtin_amdgcn_global_load_async_to_lds_b128)
#define USE_ASYNC_LDS 1
#else
#define USE_ASYNC_LDS 0
#endif

#if USE_ASYNC_LDS
typedef __attribute__((address_space(1))) v4i as1_v4i;
typedef __attribute__((address_space(3))) v4i as3_v4i;

__device__ inline void async_copy_b128(const void* g, void* l) {
    __builtin_amdgcn_global_load_async_to_lds_b128(
        (as1_v4i*)(unsigned long long)g,
        (as3_v4i*)(unsigned)(unsigned long long)l,
        0, 0);
}

#if __has_builtin(__builtin_amdgcn_s_wait_asynccnt)
#define ASYNC_WAIT(n) __builtin_amdgcn_s_wait_asynccnt(n)
#else
#define ASYNC_WAIT(n) asm volatile("s_wait_asynccnt " #n ::: "memory")
#endif
#else
#define ASYNC_WAIT(n)
#endif

// ---------------------------------------------------------------------------
// fp32 -> bf16 conversion (grid-stride over float4) — for x only
// ---------------------------------------------------------------------------
__launch_bounds__(256)
__global__ void cvt_f32_bf16(const float* __restrict__ in, bf16_t* __restrict__ out, long n4) {
    long i = (long)blockIdx.x * blockDim.x + threadIdx.x;
    long stride = (long)gridDim.x * blockDim.x;
    for (; i < n4; i += stride) {
        float4 v = ((const float4*)in)[i];
        union { bf16_t h[4]; uint2 u; } t;
        t.h[0] = (bf16_t)v.x; t.h[1] = (bf16_t)v.y;
        t.h[2] = (bf16_t)v.z; t.h[3] = (bf16_t)v.w;
        ((uint2*)out)[i] = t.u;
    }
}

// ---------------------------------------------------------------------------
// Tiled transpose + fp32->bf16: in fp32 [z][R][C] -> out bf16 [z][C][R]
// ---------------------------------------------------------------------------
__launch_bounds__(256)
__global__ void transpose_cvt_bf16(const float* __restrict__ in, bf16_t* __restrict__ out,
                                   int R, int C, long strideInZ, long strideOutZ) {
    __shared__ float tile[32][33];
    const long zi = (long)blockIdx.z * strideInZ;
    const long zo = (long)blockIdx.z * strideOutZ;
    const int tx = threadIdx.x & 31, ty = threadIdx.x >> 5;   // 32 x 8
    const int c0 = blockIdx.x * 32, r0 = blockIdx.y * 32;
#pragma unroll
    for (int j = 0; j < 4; ++j)
        tile[ty + j * 8][tx] = in[zi + (long)(r0 + ty + j * 8) * C + (c0 + tx)];
    __syncthreads();
#pragma unroll
    for (int j = 0; j < 4; ++j)
        out[zo + (long)(c0 + ty + j * 8) * R + (r0 + tx)] = (bf16_t)tile[tx][ty + j * 8];
}

// ---------------------------------------------------------------------------
// bf16 WMMA GEMM, batched over z:  C[M,N](+epi) = A[M,K] * Bt[N,K]^T
//   A : bf16 row-major [M][K]   (lda)
//   Bt: bf16 row-major [N][K]   (ldb) — "transposed-B" layout for ALL GEMMs
// Block tile 128x128, K-step 64 (2 WMMA k-chunks), 256 threads = 8 waves,
// wave tile 32x64. Double-buffered LDS fed by async global->LDS b128 copies
// (ASYNCcnt-pipelined); last iteration peeled so the steady state is
// branch-free. Batch offsets use shifts/masks (divisors are powers of two).
// ---------------------------------------------------------------------------
template<int EPI>
__launch_bounds__(256)
__global__ void gemm_wmma_bf16(
    const bf16_t* __restrict__ A, const bf16_t* __restrict__ Bt,
    const float* __restrict__ bias, const float* __restrict__ resid,
    void* __restrict__ Cout,
    int K, int lda, int ldb, int ldc,
    long aStride, int aShift,
    long bStride, int bMask,
    long cStr1, long cStr2, int cShift, int cMask,
    long biasStride, int biasMask,
    float scale)
{
    __shared__ alignas(16) bf16_t ldsA[2][128 * 72];
    __shared__ alignas(16) bf16_t ldsB[2][128 * 72];

    const int  z     = blockIdx.z;
    const long zA    = (long)(z >> aShift) * aStride;
    const long zB    = (long)(z & bMask) * bStride;
    const long zC    = (long)(z >> cShift) * cStr1 + (long)(z & cMask) * cStr2;
    const long zBias = (long)(z & biasMask) * biasStride;

    const int mBase = blockIdx.y * 128;
    const int nBase = blockIdx.x * 128;
    const int tid   = threadIdx.x;
    const int lane  = tid & 31;
    const int wave  = tid >> 5;
    const int waveM = wave & 3;    // 4 waves along M (32 rows each)
    const int waveN = wave >> 2;   // 2 waves along N (64 cols each)

    v8f acc[2][4] = {};

    // staging: thread -> (row 0..127, 32-elem k-chunk 0 or 32); 4x b128 each
    const int sr = tid >> 1;
    const int sc = (tid & 1) * 32;
    const bf16_t* gA0 = A  + zA + (long)(mBase + sr) * lda + sc;
    const bf16_t* gB0 = Bt + zB + (long)(nBase + sr) * ldb + sc;

    auto stage = [&](int kt, int buf) {
        const bf16_t* ga = gA0 + kt;
        const bf16_t* gb = gB0 + kt;
        bf16_t* la = &ldsA[buf][sr * 72 + sc];
        bf16_t* lb = &ldsB[buf][sr * 72 + sc];
#if USE_ASYNC_LDS
#pragma unroll
        for (int j = 0; j < 4; ++j) {
            async_copy_b128(ga + j * 8, la + j * 8);
            async_copy_b128(gb + j * 8, lb + j * 8);
        }
#else
#pragma unroll
        for (int j = 0; j < 4; ++j) {
            *(uint4*)(la + j * 8) = *(const uint4*)(ga + j * 8);
            *(uint4*)(lb + j * 8) = *(const uint4*)(gb + j * 8);
        }
#endif
    };

    const int half = lane >> 4;   // lane half selects k-half per ISA frag layout
    const int lm   = lane & 15;
    union Frag { v16bf v; uint4 q[2]; };

    auto compute = [&](int buf) {
#pragma unroll
        for (int kk = 0; kk < 64; kk += 32) {
            Frag af[2], bfr[4];
#pragma unroll
            for (int m = 0; m < 2; ++m) {
                const int row = waveM * 32 + m * 16 + lm;
                af[m].q[0] = *(const uint4*)&ldsA[buf][row * 72 + kk + half * 8];
                af[m].q[1] = *(const uint4*)&ldsA[buf][row * 72 + kk + 16 + half * 8];
            }
#pragma unroll
            for (int n = 0; n < 4; ++n) {
                const int col = waveN * 64 + n * 16 + lm;
                bfr[n].q[0] = *(const uint4*)&ldsB[buf][col * 72 + kk + half * 8];
                bfr[n].q[1] = *(const uint4*)&ldsB[buf][col * 72 + kk + 16 + half * 8];
            }
#pragma unroll
            for (int m = 0; m < 2; ++m)
#pragma unroll
                for (int n = 0; n < 4; ++n)
                    acc[m][n] = __builtin_amdgcn_wmma_f32_16x16x32_bf16(
                        false, af[m].v, false, bfr[n].v,
                        (short)0, acc[m][n], false, false);
        }
    };

    const int nk = K >> 6;
    stage(0, 0);
    // steady state: branch-free — always stage next tile, wait for current one
    for (int i = 0; i < nk - 1; ++i) {
        const int buf = i & 1;
        stage((i + 1) << 6, buf ^ 1);
        ASYNC_WAIT(8);           // tile i done; tile i+1 (8 ops) still in flight
        __syncthreads();
        compute(buf);
        __syncthreads();         // everyone done reading buf before restage
    }
    // peeled tail
    ASYNC_WAIT(0);
    __syncthreads();
    compute((nk - 1) & 1);

    // ---- epilogue: scale, bias, residual, relu, (transposed) store ----
#pragma unroll
    for (int m = 0; m < 2; ++m) {
#pragma unroll
        for (int n = 0; n < 4; ++n) {
            const int col = nBase + waveN * 64 + n * 16 + lm;
            float bv = 0.0f;
            if constexpr (EPI & EPI_BIAS) bv = bias[zBias + col];
#pragma unroll
            for (int g = 0; g < 8; ++g) {
                const int row = mBase + waveM * 32 + m * 16 + half * 8 + g;
                long idx;
                if constexpr (EPI & EPI_TRANSC) idx = zC + (long)col * ldc + row;
                else                            idx = zC + (long)row * ldc + col;
                float v = acc[m][n][g] * scale + bv;
                if constexpr (EPI & EPI_RESID) v += resid[idx];
                if constexpr (EPI & EPI_RELU)  v = fmaxf(v, 0.0f);
                if constexpr (EPI & EPI_OUTBF16)
                    ((bf16_t*)Cout)[idx] = (bf16_t)v;
                else
                    ((float*)Cout)[idx] = v;
            }
        }
    }
}

// ---------------------------------------------------------------------------
// Row softmax over 1024 columns: one wave per row, fp32 in -> bf16 out.
// ---------------------------------------------------------------------------
__launch_bounds__(256)
__global__ void softmax_rows_bf16(const float* __restrict__ S, bf16_t* __restrict__ P) {
    const int lane = threadIdx.x & 31;
    const int wave = threadIdx.x >> 5;
    const long row = (long)blockIdx.x * 8 + wave;
    const float4* in = (const float4*)(S + row * 1024);

    float4 f[8];
    float mx = -3.0e38f;
#pragma unroll
    for (int j = 0; j < 8; ++j) {
        f[j] = in[j * 32 + lane];
        mx = fmaxf(mx, fmaxf(fmaxf(f[j].x, f[j].y), fmaxf(f[j].z, f[j].w)));
    }
    for (int m = 16; m >= 1; m >>= 1) mx = fmaxf(mx, __shfl_xor(mx, m, 32));

    float sum = 0.0f;
#pragma unroll
    for (int j = 0; j < 8; ++j) {
        f[j].x = __expf(f[j].x - mx);
        f[j].y = __expf(f[j].y - mx);
        f[j].z = __expf(f[j].z - mx);
        f[j].w = __expf(f[j].w - mx);
        sum += f[j].x + f[j].y + f[j].z + f[j].w;
    }
    for (int m = 16; m >= 1; m >>= 1) sum += __shfl_xor(sum, m, 32);
    const float inv = 1.0f / sum;

    uint2* out = (uint2*)(P + row * 1024);
#pragma unroll
    for (int j = 0; j < 8; ++j) {
        union { bf16_t h[4]; uint2 u; } t;
        t.h[0] = (bf16_t)(f[j].x * inv);
        t.h[1] = (bf16_t)(f[j].y * inv);
        t.h[2] = (bf16_t)(f[j].z * inv);
        t.h[3] = (bf16_t)(f[j].w * inv);
        out[j * 32 + lane] = t.u;
    }
}

// ---------------------------------------------------------------------------
// LayerNorm over 512 columns: one wave per row. fp32 out (+ optional bf16).
// ---------------------------------------------------------------------------
template<bool WRITE_BF16>
__launch_bounds__(256)
__global__ void layernorm_rows(const float* __restrict__ X,
                               const float* __restrict__ g,
                               const float* __restrict__ b,
                               float* __restrict__ Y,
                               bf16_t* __restrict__ Yb) {
    const int lane = threadIdx.x & 31;
    const int wave = threadIdx.x >> 5;
    const long row = (long)blockIdx.x * 8 + wave;
    const float4* in = (const float4*)(X + row * 512);

    float4 f[4];
    float s = 0.0f, s2 = 0.0f;
#pragma unroll
    for (int j = 0; j < 4; ++j) {
        f[j] = in[j * 32 + lane];
        s  += f[j].x + f[j].y + f[j].z + f[j].w;
        s2 += f[j].x * f[j].x + f[j].y * f[j].y + f[j].z * f[j].z + f[j].w * f[j].w;
    }
    for (int m = 16; m >= 1; m >>= 1) { s += __shfl_xor(s, m, 32); s2 += __shfl_xor(s2, m, 32); }
    const float mu   = s * (1.0f / 512.0f);
    const float var  = s2 * (1.0f / 512.0f) - mu * mu;
    const float rstd = rsqrtf(var + 1e-6f);

    const float4* g4 = (const float4*)g;
    const float4* b4 = (const float4*)b;
#pragma unroll
    for (int j = 0; j < 4; ++j) {
        const int idx = j * 32 + lane;
        float4 gg = g4[idx], bb = b4[idx], y;
        y.x = (f[j].x - mu) * rstd * gg.x + bb.x;
        y.y = (f[j].y - mu) * rstd * gg.y + bb.y;
        y.z = (f[j].z - mu) * rstd * gg.z + bb.z;
        y.w = (f[j].w - mu) * rstd * gg.w + bb.w;
        ((float4*)(Y + row * 512))[idx] = y;
        if constexpr (WRITE_BF16) {
            union { bf16_t h[4]; uint2 u; } t;
            t.h[0] = (bf16_t)y.x; t.h[1] = (bf16_t)y.y;
            t.h[2] = (bf16_t)y.z; t.h[3] = (bf16_t)y.w;
            ((uint2*)(Yb + row * 512))[idx] = t.u;
        }
    }
}

// ---------------------------------------------------------------------------
// Host launcher
// ---------------------------------------------------------------------------
extern "C" void kernel_launch(void* const* d_in, const int* in_sizes, int n_in,
                              void* d_out, int out_size, void* d_ws, size_t ws_size,
                              hipStream_t stream) {
    (void)in_sizes; (void)n_in; (void)out_size; (void)ws_size;
    constexpr int B = 8, S = 1024, D = 512, H = 8, F = 2048;

    const float* x   = (const float*)d_in[0];
    const float* Wq  = (const float*)d_in[1];
    const float* bq  = (const float*)d_in[2];
    const float* Wk  = (const float*)d_in[3];
    const float* bk  = (const float*)d_in[4];
    const float* Wv  = (const float*)d_in[5];
    const float* bv  = (const float*)d_in[6];
    const float* Wo  = (const float*)d_in[7];
    const float* bo  = (const float*)d_in[8];
    const float* g1  = (const float*)d_in[9];
    const float* b1  = (const float*)d_in[10];
    const float* Wf1 = (const float*)d_in[11];
    const float* bf1 = (const float*)d_in[12];
    const float* Wf2 = (const float*)d_in[13];
    const float* bf2 = (const float*)d_in[14];
    const float* g2  = (const float*)d_in[15];
    const float* b2  = (const float*)d_in[16];
    float* out = (float*)d_out;

    // ---- workspace carve-up ----
    char* p = (char*)d_ws;
    size_t cur = 0;
    auto take = [&](size_t bytes) -> void* {
        void* r = p + cur;
        cur = (cur + bytes + 255) & ~(size_t)255;
        return r;
    };
    const long nX  = (long)B * S * D;
    const long nWh = (long)H * D * D;
    const long nQ  = (long)B * H * S * D;
    const long nSc = (long)B * H * S * S;

    bf16_t* xb    = (bf16_t*)take(nX * 2);
    bf16_t* Wqt   = (bf16_t*)take(nWh * 2);             // [H][e][d]
    bf16_t* Wkt   = (bf16_t*)take(nWh * 2);
    bf16_t* Wvt   = (bf16_t*)take(nWh * 2);
    bf16_t* Wot   = (bf16_t*)take((long)H * D * D * 2); // [512][4096]
    bf16_t* Wf1t  = (bf16_t*)take((long)D * F * 2);     // [2048][512]
    bf16_t* Wf2t  = (bf16_t*)take((long)F * D * 2);     // [512][2048]
    bf16_t* Qb    = (bf16_t*)take(nQ * 2);              // [B,H,S,D]
    bf16_t* Kb    = (bf16_t*)take(nQ * 2);              // [B,H,S,D]
    bf16_t* Vt    = (bf16_t*)take(nQ * 2);              // [B,H,D,S] (transposed)
    float*  scores = (float*)take(nSc * 4);
    bf16_t* att   = (bf16_t*)take(nSc * 2);
    bf16_t* cat   = (bf16_t*)take(nQ * 2);              // [B,S,H*D]
    float*  tmp1  = (float*)take(nX * 4);               // x + mha
    float*  hf    = (float*)take(nX * 4);               // h fp32
    bf16_t* hb    = (bf16_t*)take(nX * 2);              // h bf16
    bf16_t* f1    = (bf16_t*)take((long)B * S * F * 2);
    float*  tmp2  = (float*)take(nX * 4);               // h + ffn

    const dim3 blk(256);

    // ---- x -> bf16 (row-major, used as A everywhere) ----
    {
        long n4 = nX / 4;
        long blocks = (n4 + 255) / 256; if (blocks > 4096) blocks = 4096;
        cvt_f32_bf16<<<dim3((unsigned)blocks), blk, 0, stream>>>(x, xb, n4);
    }
    // ---- weights -> bf16 transposed ([N][K] layout for the GEMM B operand) ----
    transpose_cvt_bf16<<<dim3(D / 32, D / 32, H), blk, 0, stream>>>(
        Wq, Wqt, D, D, (long)D * D, (long)D * D);
    transpose_cvt_bf16<<<dim3(D / 32, D / 32, H), blk, 0, stream>>>(
        Wk, Wkt, D, D, (long)D * D, (long)D * D);
    transpose_cvt_bf16<<<dim3(D / 32, D / 32, H), blk, 0, stream>>>(
        Wv, Wvt, D, D, (long)D * D, (long)D * D);
    transpose_cvt_bf16<<<dim3(D / 32, (H * D) / 32, 1), blk, 0, stream>>>(
        Wo, Wot, H * D, D, 0, 0);
    transpose_cvt_bf16<<<dim3(F / 32, D / 32, 1), blk, 0, stream>>>(
        Wf1, Wf1t, D, F, 0, 0);
    transpose_cvt_bf16<<<dim3(D / 32, F / 32, 1), blk, 0, stream>>>(
        Wf2, Wf2t, F, D, 0, 0);

    // ---- Q/K projections: per (b,h) GEMM S x D x D ----
    const dim3 gQKV(D / 128, S / 128, B * H);
    gemm_wmma_bf16<EPI_BIAS | EPI_OUTBF16><<<gQKV, blk, 0, stream>>>(
        xb, Wqt, bq, nullptr, Qb, D, D, D, D,
        (long)S * D, 3, (long)D * D, 7,
        (long)S * D, 0, 0, 0, (long)D, 7, 1.0f);
    gemm_wmma_bf16<EPI_BIAS | EPI_OUTBF16><<<gQKV, blk, 0, stream>>>(
        xb, Wkt, bk, nullptr, Kb, D, D, D, D,
        (long)S * D, 3, (long)D * D, 7,
        (long)S * D, 0, 0, 0, (long)D, 7, 1.0f);
    // ---- V projection stored transposed -> Vt[B,H,D,S] ----
    gemm_wmma_bf16<EPI_BIAS | EPI_OUTBF16 | EPI_TRANSC><<<gQKV, blk, 0, stream>>>(
        xb, Wvt, bv, nullptr, Vt, D, D, D, /*ldc=*/S,
        (long)S * D, 3, (long)D * D, 7,
        (long)S * D, 0, 0, 0, (long)D, 7, 1.0f);

    // ---- scores = Q K^T / sqrt(D): B operand = K[S,D] already in [N][K] form ----
    const dim3 gSc(S / 128, S / 128, B * H);
    gemm_wmma_bf16<0><<<gSc, blk, 0, stream>>>(
        Qb, Kb, nullptr, nullptr, scores, D, D, D, S,
        (long)S * D, 0, (long)S * D, 63,
        (long)S * S, 0, 0, 0, 0, 0,
        0.044194173824159216f /* 1/sqrt(512) */);

    // ---- softmax rows (B*H*S = 65536 rows of 1024) ----
    softmax_rows_bf16<<<dim3((B * H * S) / 8), blk, 0, stream>>>(scores, att);

    // ---- O = att * V: B operand = Vt[D,S] in [N][K] form; C scattered to cat ----
    const dim3 gPV(D / 128, S / 128, B * H);
    gemm_wmma_bf16<EPI_OUTBF16><<<gPV, blk, 0, stream>>>(
        att, Vt, nullptr, nullptr, cat, S, S, S, H * D,
        (long)S * S, 0, (long)S * D, 63,
        (long)S * H * D, (long)D, 3, 7, 0, 0, 1.0f);

    // ---- output projection + residual: [8192 x 4096] x [4096 x 512] + x ----
    const dim3 gO(D / 128, (B * S) / 128, 1);
    gemm_wmma_bf16<EPI_BIAS | EPI_RESID><<<gO, blk, 0, stream>>>(
        cat, Wot, bo, x, tmp1, H * D, H * D, H * D, D,
        0, 0, 0, 0, 0, 0, 0, 0, 0, 0, 1.0f);

    // ---- LN1 -> h (fp32 + bf16) ----
    layernorm_rows<true><<<dim3((B * S) / 8), blk, 0, stream>>>(tmp1, g1, b1, hf, hb);

    // ---- FFN1 + ReLU: [8192 x 512] x [512 x 2048] ----
    const dim3 gF1(F / 128, (B * S) / 128, 1);
    gemm_wmma_bf16<EPI_BIAS | EPI_RELU | EPI_OUTBF16><<<gF1, blk, 0, stream>>>(
        hb, Wf1t, bf1, nullptr, f1, D, D, D, F,
        0, 0, 0, 0, 0, 0, 0, 0, 0, 0, 1.0f);

    // ---- FFN2 + residual: [8192 x 2048] x [2048 x 512] + h ----
    const dim3 gF2(D / 128, (B * S) / 128, 1);
    gemm_wmma_bf16<EPI_BIAS | EPI_RESID><<<gF2, blk, 0, stream>>>(
        f1, Wf2t, bf2, hf, tmp2, F, F, F, D,
        0, 0, 0, 0, 0, 0, 0, 0, 0, 0, 1.0f);

    // ---- LN2 -> final output ----
    layernorm_rows<false><<<dim3((B * S) / 8), blk, 0, stream>>>(tmp2, g2, b2, out, nullptr);
}
